// CasualAttention_80564996538885
// MI455X (gfx1250) — matrix operand order repaired and checked by reading, not compiled, not run on hardware
//
#include <hip/hip_runtime.h>
#include <hip/hip_bf16.h>
#include <math.h>

typedef __bf16 bf16_t;
typedef __bf16 v16bf __attribute__((ext_vector_type(16)));
typedef __bf16 v8bf  __attribute__((ext_vector_type(8)));
typedef __bf16 v2bf  __attribute__((ext_vector_type(2)));
typedef float  v8f   __attribute__((ext_vector_type(8)));
typedef int    v4i_t __attribute__((vector_size(16)));

#define B_SZ   2
#define C_SEQ  2048
#define H_DIM  1024
#define NHEAD  16
#define HDH    64
#define M_TOT  (B_SZ * C_SEQ)   // 4096
#define N_QKV  (3 * H_DIM)      // 3072

// ---------------------------------------------------------------------------
// CDNA5 async global->LDS copy (ASYNCcnt path), with synchronous fallback.
// Builtin signature (from hipcc diagnostic): (v4i AS1*, v4i AS3*, Ii, Ii).
// ---------------------------------------------------------------------------
#if defined(__gfx1250__) && __has_builtin(__builtin_amdgcn_global_load_async_to_lds_b128)
#define HAVE_ASYNC_LDS 1
#else
#define HAVE_ASYNC_LDS 0
#endif

__device__ __forceinline__ void copy16_g2l(const bf16_t* g, bf16_t* l) {
#if HAVE_ASYNC_LDS
  __builtin_amdgcn_global_load_async_to_lds_b128(
      (__attribute__((address_space(1))) v4i_t*)(void*)g,
      (__attribute__((address_space(3))) v4i_t*)(void*)l,
      /*offset=*/0, /*cpol=*/0);
#else
  *(v8bf*)l = *(const v8bf*)g;
#endif
}

__device__ __forceinline__ void async_wait_all() {
#if HAVE_ASYNC_LDS
#if __has_builtin(__builtin_amdgcn_s_wait_asynccnt)
  __builtin_amdgcn_s_wait_asynccnt(0);
#else
  asm volatile("s_wait_asynccnt 0" ::: "memory");
#endif
#endif
}

// ---------------------------------------------------------------------------
// WMMA helpers (CDNA5 wave32, 16x16x32 bf16 -> f32)
// ---------------------------------------------------------------------------
__device__ __forceinline__ v8f wmma_bf16f32(v16bf a, v16bf b, v8f c) {
  return __builtin_amdgcn_wmma_f32_16x16x32_bf16(false, a, false, b,
                                                 (short)0, c, false, false);
}

// A-matrix 16x32 (MxK) bf16 layout: lane<16 -> row M=lane, elems K=k0..k0+7
// then k0+16..k0+23; lane>=16 -> row M=lane-16, K=k0+8..15 then k0+24..31.
__device__ __forceinline__ v16bf loadA16x32(const bf16_t* base, int ld,
                                            int m0, int k0, int lane) {
  int row = m0 + (lane & 15);
  int kb  = k0 + ((lane & 16) ? 8 : 0);
  const bf16_t* p = base + (size_t)row * ld + kb;
  v8bf lo = *(const v8bf*)p;
  v8bf hi = *(const v8bf*)(p + 16);
  return __builtin_shufflevector(lo, hi, 0,1,2,3,4,5,6,7,8,9,10,11,12,13,14,15);
}

// B-matrix 32x16 (KxN) bf16, stored N-major (base[n*ld + k]): lane<16 holds
// column n0+lane, K=k0..k0+7 then k0+16..23; lane>=16 column n0+lane-16, K=k0+8..15 then k0+24..31.
__device__ __forceinline__ v16bf loadB32x16(const bf16_t* base, int ld,
                                            int k0, int n0, int lane) {
  int col = n0 + (lane & 15);
  int kb  = k0 + ((lane & 16) ? 8 : 0);
  const bf16_t* p = base + (size_t)col * ld + kb;
  v8bf lo = *(const v8bf*)p;
  v8bf hi = *(const v8bf*)(p + 16);
  return __builtin_shufflevector(lo, hi, 0,1,2,3,4,5,6,7,8,9,10,11,12,13,14,15);
}

// ---------------------------------------------------------------------------
// fp32 -> bf16 conversion (grid-stride)
// ---------------------------------------------------------------------------
__global__ void cvt_f32_bf16(const float* __restrict__ in,
                             bf16_t* __restrict__ out, int n) {
  for (int i = blockIdx.x * blockDim.x + threadIdx.x; i < n;
       i += gridDim.x * blockDim.x)
    out[i] = (bf16_t)in[i];
}

// ---------------------------------------------------------------------------
// QKV projection: qkv[m,n] = sum_k x[m,k]*Wqkv[n,k] + b[n]; Q gets *0.125.
// Wave: 32(M) x 64(N); block 8 waves = 64 x 256. Grid (64, 12).
// ---------------------------------------------------------------------------
__global__ __launch_bounds__(256) void qkv_gemm(const bf16_t* __restrict__ X,
                                                const bf16_t* __restrict__ W,
                                                const float* __restrict__ bias,
                                                bf16_t* __restrict__ qkv) {
  const int lane = threadIdx.x & 31;
  const int wave = threadIdx.x >> 5;
  const int m0 = blockIdx.x * 64 + (wave & 1) * 32;
  const int n0 = blockIdx.y * 256 + (wave >> 1) * 64;

  v8f acc[2][4] = {};
  for (int k0 = 0; k0 < H_DIM; k0 += 32) {
    v16bf a0 = loadA16x32(X, H_DIM, m0,      k0, lane);
    v16bf a1 = loadA16x32(X, H_DIM, m0 + 16, k0, lane);
#pragma unroll
    for (int j = 0; j < 4; ++j) {
      v16bf bt = loadB32x16(W, H_DIM, k0, n0 + 16 * j, lane);
      acc[0][j] = wmma_bf16f32(a0, bt, acc[0][j]);
      acc[1][j] = wmma_bf16f32(a1, bt, acc[1][j]);
    }
  }
  const int nc = lane & 15;
  const int mo = (lane & 16) ? 8 : 0;
#pragma unroll
  for (int j = 0; j < 4; ++j) {
    int n = n0 + 16 * j + nc;
    float bv = bias[n];
    float sc = (n < H_DIM) ? 0.125f : 1.0f;  // fold attention scale into Q
#pragma unroll
    for (int i = 0; i < 2; ++i) {
#pragma unroll
      for (int r = 0; r < 8; ++r) {
        int m = m0 + 16 * i + mo + r;
        qkv[(size_t)m * N_QKV + n] = (bf16_t)((acc[i][j][r] + bv) * sc);
      }
    }
  }
}

// ---------------------------------------------------------------------------
// Flash attention. Block = 4 waves = 64 queries of one (b,h). Grid (32, 32).
// Double-buffered K/V staging in LDS, one barrier per key tile.
// ---------------------------------------------------------------------------
__global__ __launch_bounds__(128) void attn_kernel(const bf16_t* __restrict__ qkv,
                                                   bf16_t* __restrict__ Y) {
  __shared__ alignas(32) bf16_t Kl[2][32 * 64];    // [key][d]
  __shared__ alignas(32) bf16_t Vt[2][64 * 32];    // [d][key] (transposed)
  __shared__ alignas(32) bf16_t Pl[4][16 * 32];    // per-wave P bounce

  const int tid  = threadIdx.x;
  const int lane = tid & 31;
  const int wave = tid >> 5;
  const int b = blockIdx.y / NHEAD;
  const int h = blockIdx.y % NHEAD;
  const int qb = blockIdx.x * 64;
  const int q0 = qb + wave * 16;

  const bf16_t* base = qkv + (size_t)b * C_SEQ * N_QKV + h * HDH;
  const bf16_t* Qb = base;
  const bf16_t* Kb = base + H_DIM;
  const bf16_t* Vb = base + 2 * H_DIM;

  // K tile staging: thread -> (key, 32B segment); async global->LDS when available.
  auto stageK = [&](int kt0, int buf) {
    int key = tid & 31, seg = tid >> 5;          // seg 0..3 (16 elems each)
    const bf16_t* src = Kb + (size_t)(kt0 + key) * N_QKV + seg * 16;
    bf16_t* dst = &Kl[buf][key * 64 + seg * 16];
    copy16_g2l(src, dst);
    copy16_g2l(src + 8, dst + 8);
  };
  // V tile staging (transposed): thread -> key pair (2c, 2c+1) x 8 d-values,
  // packed b32 stores (conflict-free: lanes cover 16 distinct banks per seg).
  auto stageV = [&](int kt0, int buf) {
    int c = tid & 15, seg = tid >> 4;            // seg 0..7 (8 d each)
    int d0 = seg * 8;
    const bf16_t* r0 = Vb + (size_t)(kt0 + 2 * c) * N_QKV + d0;
    v8bf va = *(const v8bf*)r0;
    v8bf vb = *(const v8bf*)(r0 + N_QKV);
#pragma unroll
    for (int e = 0; e < 8; ++e) {
      v2bf pk; pk[0] = va[e]; pk[1] = vb[e];
      *(v2bf*)&Vt[buf][(d0 + e) * 32 + 2 * c] = pk;
    }
  };

  // Q tile (16 queries x 64 d) resident in A-layout registers, pre-scaled.
  v16bf aq0 = loadA16x32(Qb, N_QKV, q0, 0, lane);
  v16bf aq1 = loadA16x32(Qb, N_QKV, q0, 32, lane);

  float mrow[8], lrow[8];
  v8f o[4] = {};
#pragma unroll
  for (int r = 0; r < 8; ++r) { mrow[r] = -INFINITY; lrow[r] = 0.f; }

  const int nc = lane & 15;
  const int mo = (lane & 16) ? 8 : 0;
  const int ktiles = (qb + 64) >> 5;   // causal upper bound; always >= 2

  stageK(0, 0);
  stageV(0, 0);
  async_wait_all();
  __syncthreads();

  for (int kt = 0; kt < ktiles; ++kt) {
    const int kt0 = kt << 5;
    const int buf = kt & 1;

    // prefetch next tile into the other buffer while this one is consumed
    if (kt + 1 < ktiles) {
      stageK((kt + 1) << 5, buf ^ 1);
      stageV((kt + 1) << 5, buf ^ 1);
    }

    if (kt0 <= q0 + 15) {  // wave-uniform: skip fully-masked tiles
      const bf16_t* Kb_l = &Kl[buf][0];
      const bf16_t* Vt_l = &Vt[buf][0];
      v8f s0 = {}, s1 = {};
      s0 = wmma_bf16f32(aq0, loadB32x16(Kb_l, 64, 0,  0,  lane), s0);
      s1 = wmma_bf16f32(aq0, loadB32x16(Kb_l, 64, 0,  16, lane), s1);
      s0 = wmma_bf16f32(aq1, loadB32x16(Kb_l, 64, 32, 0,  lane), s0);
      s1 = wmma_bf16f32(aq1, loadB32x16(Kb_l, 64, 32, 16, lane), s1);

      if (kt0 + 31 > q0) {  // diagonal tile: causal mask
#pragma unroll
        for (int r = 0; r < 8; ++r) {
          int q = q0 + r + mo;
          if (kt0 + nc      > q) s0[r] = -INFINITY;
          if (kt0 + 16 + nc > q) s1[r] = -INFINITY;
        }
      }

      // online softmax (row = accumulator element; reduce over 16-lane half)
      float p0[8], p1[8];
#pragma unroll
      for (int r = 0; r < 8; ++r) {
        float tmax = fmaxf(s0[r], s1[r]);
        tmax = fmaxf(tmax, __shfl_xor(tmax, 1));
        tmax = fmaxf(tmax, __shfl_xor(tmax, 2));
        tmax = fmaxf(tmax, __shfl_xor(tmax, 4));
        tmax = fmaxf(tmax, __shfl_xor(tmax, 8));
        float mnew  = fmaxf(mrow[r], tmax);
        float alpha = __expf(mrow[r] - mnew);
        p0[r] = __expf(s0[r] - mnew);
        p1[r] = __expf(s1[r] - mnew);
        float ts = p0[r] + p1[r];
        ts += __shfl_xor(ts, 1);
        ts += __shfl_xor(ts, 2);
        ts += __shfl_xor(ts, 4);
        ts += __shfl_xor(ts, 8);
        lrow[r] = lrow[r] * alpha + ts;
        mrow[r] = mnew;
#pragma unroll
        for (int j = 0; j < 4; ++j) o[j][r] *= alpha;
      }

      // accumulator layout -> A layout via per-wave LDS bounce
      bf16_t* pw = &Pl[wave][0];
#pragma unroll
      for (int r = 0; r < 8; ++r) {
        pw[(r + mo) * 32 + nc]      = (bf16_t)p0[r];
        pw[(r + mo) * 32 + 16 + nc] = (bf16_t)p1[r];
      }
      {
        int qr = lane & 15;
        int ko = (lane & 16) ? 8 : 0;
        const bf16_t* pp = pw + qr * 32 + ko;
        v8bf lo = *(const v8bf*)pp;
        v8bf hi = *(const v8bf*)(pp + 16);
        v16bf aP = __builtin_shufflevector(lo, hi,
                     0,1,2,3,4,5,6,7,8,9,10,11,12,13,14,15);
#pragma unroll
        for (int j = 0; j < 4; ++j) {
          v16bf bV = loadB32x16(Vt_l, 32, 0, j * 16, lane);
          o[j] = wmma_bf16f32(aP, bV, o[j]);
        }
      }
    }

    async_wait_all();   // our async LDS writes must land before we signal
    __syncthreads();
  }

  float invl[8];
#pragma unroll
  for (int r = 0; r < 8; ++r) invl[r] = 1.0f / lrow[r];
#pragma unroll
  for (int j = 0; j < 4; ++j)
#pragma unroll
    for (int r = 0; r < 8; ++r) {
      int q = q0 + r + mo;
      Y[(size_t)(b * C_SEQ + q) * H_DIM + h * HDH + j * 16 + nc] =
          (bf16_t)(o[j][r] * invl[r]);
    }
}

// ---------------------------------------------------------------------------
// Output projection: out[m,n] = sum_k Y[m,k]*Wo[n,k] + bo[n] (fp32 out).
// Grid (64, 4), block 256.
// ---------------------------------------------------------------------------
__global__ __launch_bounds__(256) void out_gemm(const bf16_t* __restrict__ Yin,
                                                const bf16_t* __restrict__ W,
                                                const float* __restrict__ bias,
                                                float* __restrict__ out) {
  const int lane = threadIdx.x & 31;
  const int wave = threadIdx.x >> 5;
  const int m0 = blockIdx.x * 64 + (wave & 1) * 32;
  const int n0 = blockIdx.y * 256 + (wave >> 1) * 64;

  v8f acc[2][4] = {};
  for (int k0 = 0; k0 < H_DIM; k0 += 32) {
    v16bf a0 = loadA16x32(Yin, H_DIM, m0,      k0, lane);
    v16bf a1 = loadA16x32(Yin, H_DIM, m0 + 16, k0, lane);
#pragma unroll
    for (int j = 0; j < 4; ++j) {
      v16bf bt = loadB32x16(W, H_DIM, k0, n0 + 16 * j, lane);
      acc[0][j] = wmma_bf16f32(a0, bt, acc[0][j]);
      acc[1][j] = wmma_bf16f32(a1, bt, acc[1][j]);
    }
  }
  const int nc = lane & 15;
  const int mo = (lane & 16) ? 8 : 0;
#pragma unroll
  for (int j = 0; j < 4; ++j) {
    int n = n0 + 16 * j + nc;
    float bv = bias[n];
#pragma unroll
    for (int i = 0; i < 2; ++i)
#pragma unroll
      for (int r = 0; r < 8; ++r) {
        int m = m0 + 16 * i + mo + r;
        out[(size_t)m * H_DIM + n] = acc[i][j][r] + bv;
      }
  }
}

// ---------------------------------------------------------------------------
extern "C" void kernel_launch(void* const* d_in, const int* in_sizes, int n_in,
                              void* d_out, int out_size, void* d_ws, size_t ws_size,
                              hipStream_t stream) {
  (void)in_sizes; (void)n_in; (void)out_size; (void)ws_size;
  const float* x    = (const float*)d_in[0];
  const float* wqkv = (const float*)d_in[1];
  const float* bqkv = (const float*)d_in[2];
  const float* wo   = (const float*)d_in[3];
  const float* bo   = (const float*)d_in[4];
  float* out = (float*)d_out;

  char* ws = (char*)d_ws;
  size_t off = 0;
  auto carve = [&](size_t bytes) -> void* {
    void* p = ws + off;
    off += (bytes + 255) & ~(size_t)255;
    return p;
  };
  bf16_t* xb   = (bf16_t*)carve((size_t)M_TOT * H_DIM * 2);  // 8 MB
  bf16_t* wqb  = (bf16_t*)carve((size_t)N_QKV * H_DIM * 2);  // 6 MB
  bf16_t* wob  = (bf16_t*)carve((size_t)H_DIM * H_DIM * 2);  // 2 MB
  bf16_t* qkvb = (bf16_t*)carve((size_t)M_TOT * N_QKV * 2);  // 24 MB
  bf16_t* yb   = (bf16_t*)carve((size_t)M_TOT * H_DIM * 2);  // 8 MB

  cvt_f32_bf16<<<1024, 256, 0, stream>>>(x,    xb,  M_TOT * H_DIM);
  cvt_f32_bf16<<<1024, 256, 0, stream>>>(wqkv, wqb, N_QKV * H_DIM);
  cvt_f32_bf16<<<1024, 256, 0, stream>>>(wo,   wob, H_DIM * H_DIM);

  qkv_gemm<<<dim3(M_TOT / 64, N_QKV / 256), 256, 0, stream>>>(xb, wqb, bqkv, qkvb);
  attn_kernel<<<dim3(C_SEQ / 64, B_SZ * NHEAD), 128, 0, stream>>>(qkvb, yb);
  out_gemm<<<dim3(M_TOT / 64, H_DIM / 256), 256, 0, stream>>>(yb, wob, bo, out);
}
